// PointNetWithRayBackbone_74783970558460
// MI455X (gfx1250) — matrix operand order repaired and checked by reading, not compiled, not run on hardware
//
#include <hip/hip_runtime.h>
#include <hip/hip_bf16.h>

// ---------------------------------------------------------------------------
// CDNA5 (gfx1250) implementation of the PointNet-with-ray 3D conv backbone.
// All convolutions run as implicit GEMM on v_wmma_f32_16x16x32_bf16.
// Activations: NDHWC bf16 (stored as u16), channels padded to multiples of 32.
// Weights: repacked on device into the exact wave32 B-matrix lane layout.
// Halo handling is branch-free: invalid lanes load from a zeroed page.
// Tile shapes: <4,1> for Cout=16 layers, <4,2> for Cout>=32 layers
// (minimizes L2 bytes per wmma: 1.25 KB and 0.75 KB respectively).
// ---------------------------------------------------------------------------

typedef __attribute__((ext_vector_type(16))) __bf16       v16bf;
typedef __attribute__((ext_vector_type(8)))  float        v8f;
typedef __attribute__((ext_vector_type(4)))  unsigned int u32x4;

union ABfrag {
    v16bf v;
    u32x4 q[2];
};

__device__ __forceinline__ unsigned short f2bf_u(float f) {
    unsigned u = __float_as_uint(f);
    unsigned r = u + 0x7FFFu + ((u >> 16) & 1u);   // round-to-nearest-even
    return (unsigned short)(r >> 16);
}
__device__ __forceinline__ float bf2f(unsigned short s) {
    return __uint_as_float(((unsigned)s) << 16);
}

// ---------------------------------------------------------------------------
// Zero fill (u32 granularity; every region is a multiple of 4 bytes).
// ---------------------------------------------------------------------------
__global__ void zero_u32(unsigned int* __restrict__ p, long n) {
    long i = (long)blockIdx.x * blockDim.x + threadIdx.x;
    long stride = (long)gridDim.x * blockDim.x;
    for (; i < n; i += stride) p[i] = 0u;
}

// ---------------------------------------------------------------------------
// Weight repack: DHWIO f32 -> per-lane bf16 B-matrix tiles.
// Layout: [tap][kc][ntile][lane(32)][16 bf16]
//   lane <16 : K = kc*32 + (j<8 ? j : 8+j)    (j = 0..15, 2 per dword), N = lane
//   lane >=16: K = kc*32 + 8 + (j<8 ? j : 8+j),                        N = lane-16
// ---------------------------------------------------------------------------
__global__ void pack_weights(const float* __restrict__ W,
                             unsigned short* __restrict__ wpk,
                             int ntaps, int Cin, int Cinp, int Cout) {
    int KC = Cinp >> 5;
    int Ntiles = Cout >> 4;
    long total = (long)ntaps * KC * Ntiles * 32 * 16;
    long i = (long)blockIdx.x * blockDim.x + threadIdx.x;
    if (i >= total) return;
    int j    = (int)(i & 15);
    int lane = (int)((i >> 4) & 31);
    long t   = i >> 9;
    int ntile = (int)(t % Ntiles);  t /= Ntiles;
    int kc    = (int)(t % KC);
    int tap   = (int)(t / KC);
    int k = kc * 32 + ((lane >= 16) ? 8 : 0) + ((j < 8) ? j : 8 + j);
    int n = ntile * 16 + (lane & 15);
    float v = (k < Cin) ? W[((long)tap * Cin + k) * Cout + n] : 0.0f;
    wpk[i] = f2bf_u(v);
}

// ---------------------------------------------------------------------------
// Ray embedding + scatter into dense bf16 grid (65 ch, padded to 96) + mask.
// ---------------------------------------------------------------------------
__global__ void embed_scatter(const float* __restrict__ vf,
                              const int* __restrict__ vc,
                              unsigned short* __restrict__ g0,
                              float* __restrict__ m0,
                              int N, int Dg, int Hg, int Wg, int C0p) {
    int i = blockIdx.x * blockDim.x + threadIdx.x;
    if (i >= N) return;
    float f0 = vf[i * 5 + 0], f1 = vf[i * 5 + 1], f2 = vf[i * 5 + 2];
    float f3 = vf[i * 5 + 3], f4 = vf[i * 5 + 4];
    int b = vc[i * 4 + 0], z = vc[i * 4 + 1], y = vc[i * 4 + 2], x = vc[i * 4 + 3];
    long vox = (((long)b * Dg + z) * Hg + y) * Wg + x;
    unsigned short* p = g0 + vox * C0p;
    float dist = sqrtf(f0 * f0 + f1 * f1 + f2 * f2);
    float inv  = 1.0f / (dist + 1e-6f);
    float dir[3] = { f0 * inv, f1 * inv, f2 * inv };
    p[0] = f2bf_u(f0); p[1] = f2bf_u(f1); p[2] = f2bf_u(f2);
    p[3] = f2bf_u(f3); p[4] = f2bf_u(f4);
    for (int a = 0; a < 3; ++a) {
        for (int fq = 0; fq < 10; ++fq) {
            float e = dir[a] * (float)(1 << fq);
            int idx = a * 10 + fq;
            p[5  + idx] = f2bf_u(sinf(e));
            p[35 + idx] = f2bf_u(cosf(e));
        }
    }
    m0[vox] = 1.0f;
}

// ---------------------------------------------------------------------------
// Mask down-sample: new_mask = (conv(mask, ones(kd,kh,kw), stride, pad) > 0)
// ---------------------------------------------------------------------------
__global__ void mask_down(const float* __restrict__ mi, float* __restrict__ mo,
                          int B, int Din, int Hin, int Win,
                          int Dout, int Hout, int Wout,
                          int kd, int kh, int kw,
                          int sd, int sh, int sw,
                          int pd, int ph, int pw) {
    int i = blockIdx.x * blockDim.x + threadIdx.x;
    int nvox = B * Dout * Hout * Wout;
    if (i >= nvox) return;
    int w = i % Wout; int t = i / Wout;
    int h = t % Hout; t /= Hout;
    int d = t % Dout; int b = t / Dout;
    int id0 = d * sd - pd, ih0 = h * sh - ph, iw0 = w * sw - pw;
    float acc = 0.f;
    for (int tz = 0; tz < kd; ++tz) {
        int id = id0 + tz; if ((unsigned)id >= (unsigned)Din) continue;
        for (int ty = 0; ty < kh; ++ty) {
            int ih = ih0 + ty; if ((unsigned)ih >= (unsigned)Hin) continue;
            for (int tx = 0; tx < kw; ++tx) {
                int iw = iw0 + tx; if ((unsigned)iw >= (unsigned)Win) continue;
                acc += mi[(((long)b * Din + id) * Hin + ih) * Win + iw];
            }
        }
    }
    mo[i] = (acc > 0.f) ? 1.f : 0.f;
}

// ---------------------------------------------------------------------------
// Implicit-GEMM conv3d on WMMA bf16.
// One wave computes an (MT*16 voxels) x (NT*16 out-channels) tile:
// each A fragment is reused by NT wmmas, each B fragment by MT wmmas.
// Halo loads are branch-free via a zeroed 256-B page (zpad).
// Epilogue: relu(acc*scale + bias [+ residual]) * mask  ->  bf16 NDHWC out.
// ---------------------------------------------------------------------------
template <int MT, int NT>
__global__ void __launch_bounds__(128)
conv3d_wmma(const unsigned short* __restrict__ in,
            const unsigned short* __restrict__ wpk,
            const float* __restrict__ scale,
            const float* __restrict__ bias,
            const float* __restrict__ maskOut,
            const unsigned short* __restrict__ residual,  // nullptr if none
            const unsigned short* __restrict__ zpad,      // 256 B of zeros
            unsigned short* __restrict__ out,
            int B, int Din, int Hin, int Win, int Cinp,
            int Dout, int Hout, int Wout, int Coutp, int Cout,
            int kd, int kh, int kw,
            int sd, int sh, int sw,
            int pd, int ph, int pw) {
    const int lane   = threadIdx.x & 31;
    const int waveId = threadIdx.x >> 5;
    const int nvox   = B * Dout * Hout * Wout;
    const int Msuper = (nvox + 16 * MT - 1) / (16 * MT);
    const int Nsuper = Cout / (16 * NT);
    const int Ntiles = Cout >> 4;
    const int tile   = blockIdx.x * 4 + waveId;
    if (tile >= Msuper * Nsuper) return;
    const int ms = tile / Nsuper;
    const int ns = tile % Nsuper;
    const int ntbase = ns * NT;

    const int mrow = lane & 15;             // A row owned by this lane
    const int koff = (lane < 16) ? 0 : 8;   // channel offset of lane's K half
    const int KC   = Cinp >> 5;
    const int HWin = Hin * Win;

    // Per-m-subtile voxel decomposition (A rows).
    int  id0[MT], ih0[MT], iw0[MT];
    long vbase[MT];
    bool rowValid[MT];
#pragma unroll
    for (int mi = 0; mi < MT; ++mi) {
        int vm = (ms * MT + mi) * 16 + mrow;
        rowValid[mi] = (vm < nvox);
        int vmc = rowValid[mi] ? vm : 0;
        int w = vmc % Wout; int t = vmc / Wout;
        int h = t % Hout;   t /= Hout;
        int d = t % Dout;   int b = t / Dout;
        id0[mi] = d * sd - pd;
        ih0[mi] = h * sh - ph;
        iw0[mi] = w * sw - pw;
        vbase[mi] = (long)b * Din * HWin;
    }

    v8f acc[MT][NT];
    const v8f vzero = {0.f, 0.f, 0.f, 0.f, 0.f, 0.f, 0.f, 0.f};
#pragma unroll
    for (int mi = 0; mi < MT; ++mi)
#pragma unroll
        for (int ni = 0; ni < NT; ++ni) acc[mi][ni] = vzero;

    // Weight pointer: affine in the (tap,kc) step -> single incremented ptr.
    const unsigned short* wp = wpk + ((long)ntbase * 32 + lane) * 16;
    const long wstep = (long)Ntiles * 512;

    for (int tz = 0; tz < kd; ++tz) {
        for (int ty = 0; ty < kh; ++ty) {
            for (int tx = 0; tx < kw; ++tx) {
                const unsigned short* ap[MT];
#pragma unroll
                for (int mi = 0; mi < MT; ++mi) {
                    const int id = id0[mi] + tz;
                    const int ih = ih0[mi] + ty;
                    const int iw = iw0[mi] + tx;
                    const bool valid = rowValid[mi] &&
                        (unsigned)id < (unsigned)Din &&
                        (unsigned)ih < (unsigned)Hin &&
                        (unsigned)iw < (unsigned)Win;
                    const long base =
                        (vbase[mi] + (long)id * HWin + (long)ih * Win + iw) * Cinp + koff;
                    ap[mi] = valid ? (in + base) : zpad;   // branch-free halo
                }
                for (int kc = 0; kc < KC; ++kc) {
                    ABfrag a[MT], bm[NT];
#pragma unroll
                    for (int mi = 0; mi < MT; ++mi) {
                        const u32x4* pa = (const u32x4*)(ap[mi] + kc * 32);
                        a[mi].q[0] = pa[0];    // channels koff..koff+7
                        a[mi].q[1] = pa[2];    // channels koff+16..koff+23
                    }
#pragma unroll
                    for (int ni = 0; ni < NT; ++ni) {
                        const u32x4* pb = (const u32x4*)(wp + ni * 512);
                        bm[ni].q[0] = pb[0];
                        bm[ni].q[1] = pb[1];
                    }
#pragma unroll
                    for (int mi = 0; mi < MT; ++mi)
#pragma unroll
                        for (int ni = 0; ni < NT; ++ni)
                            acc[mi][ni] = __builtin_amdgcn_wmma_f32_16x16x32_bf16(
                                false, a[mi].v, false, bm[ni].v,
                                (short)0, acc[mi][ni], false, false);
                    wp += wstep;
                }
            }
        }
    }

    // Epilogue. C/D layout: lanes 0-15 -> N=lane, rows M=0..7 in VGPR 0..7;
    // lanes 16-31 -> N=lane-16, rows M=8..15.
#pragma unroll
    for (int ni = 0; ni < NT; ++ni) {
        const int n  = (ntbase + ni) * 16 + (lane & 15);
        const float sc = scale[n];
        const float bi = bias[n];
#pragma unroll
        for (int mi = 0; mi < MT; ++mi) {
            const int mbase = (ms * MT + mi) * 16 + ((lane < 16) ? 0 : 8);
#pragma unroll
            for (int r = 0; r < 8; ++r) {
                const int vr = mbase + r;
                if (vr < nvox) {
                    float v = acc[mi][ni][r] * sc + bi;
                    if (residual) v += bf2f(residual[(long)vr * Coutp + n]);
                    v = fmaxf(v, 0.0f) * maskOut[vr];
                    out[(long)vr * Coutp + n] = f2bf_u(v);
                }
            }
        }
    }
}

// ---------------------------------------------------------------------------
// Final layout change: (2,2,8,8,128) bf16 -> (2, 128*2, 8, 8) f32.
// out[b][c*2+d][h][w] = h[b,d,h,w,c]
// ---------------------------------------------------------------------------
__global__ void final_transpose(const unsigned short* __restrict__ ex,
                                float* __restrict__ out) {
    int i = blockIdx.x * blockDim.x + threadIdx.x;
    if (i >= 2 * 256 * 8 * 8) return;
    int w  = i & 7;
    int h  = (i >> 3) & 7;
    int cd = (i >> 6) & 255;
    int b  = i >> 14;
    int c  = cd >> 1;
    int d  = cd & 1;
    out[i] = bf2f(ex[((((long)(b * 2 + d) * 8 + h) * 8 + w) * 128) + c]);
}

// ---------------------------------------------------------------------------
// Host-side conv launcher (template dispatch on tile shape).
// ---------------------------------------------------------------------------
template <int MT, int NT>
static void launch_conv(hipStream_t stream,
                        const unsigned short* in_, const unsigned short* wpk,
                        const float* sc, const float* bi, const float* mask,
                        const unsigned short* res, const unsigned short* zpad,
                        unsigned short* out_, int Bn,
                        int Din, int Hin, int Win, int Cinp,
                        int Dout, int Hout, int Wout, int Coutp, int Cout,
                        int kd, int kh, int kw, int sd, int sh, int sw,
                        int pd, int ph, int pw) {
    int nvox  = Bn * Dout * Hout * Wout;
    int tiles = ((nvox + 16 * MT - 1) / (16 * MT)) * (Cout / (16 * NT));
    int blocks = (tiles + 3) / 4;
    conv3d_wmma<MT, NT><<<blocks, 128, 0, stream>>>(
        in_, wpk, sc, bi, mask, res, zpad, out_,
        Bn, Din, Hin, Win, Cinp, Dout, Hout, Wout, Coutp, Cout,
        kd, kh, kw, sd, sh, sw, pd, ph, pw);
}

// ---------------------------------------------------------------------------
// Host orchestration (graph-capture safe: only launches on `stream`).
// ---------------------------------------------------------------------------
extern "C" void kernel_launch(void* const* d_in, const int* in_sizes, int n_in,
                              void* d_out, int out_size, void* d_ws, size_t ws_size,
                              hipStream_t stream) {
    (void)n_in; (void)out_size; (void)ws_size;

    const float* vf = (const float*)d_in[0];
    const int*   vc = (const int*)d_in[1];
    const int N = in_sizes[0] / 5;

    // Parameter leaves in setup_inputs() insertion order.
    const float* W_in  = (const float*)d_in[4];
    const float* S_in  = (const float*)d_in[5];
    const float* B_in  = (const float*)d_in[6];
    const float* b1a_w1 = (const float*)d_in[7];  const float* b1a_s1 = (const float*)d_in[8];  const float* b1a_b1 = (const float*)d_in[9];
    const float* b1a_w2 = (const float*)d_in[10]; const float* b1a_s2 = (const float*)d_in[11]; const float* b1a_b2 = (const float*)d_in[12];
    const float* b1b_w1 = (const float*)d_in[13]; const float* b1b_s1 = (const float*)d_in[14]; const float* b1b_b1 = (const float*)d_in[15];
    const float* b1b_w2 = (const float*)d_in[16]; const float* b1b_s2 = (const float*)d_in[17]; const float* b1b_b2 = (const float*)d_in[18];
    const float* d2_w = (const float*)d_in[19]; const float* d2_s = (const float*)d_in[20]; const float* d2_b = (const float*)d_in[21];
    const float* b2a_w1 = (const float*)d_in[22]; const float* b2a_s1 = (const float*)d_in[23]; const float* b2a_b1 = (const float*)d_in[24];
    const float* b2a_w2 = (const float*)d_in[25]; const float* b2a_s2 = (const float*)d_in[26]; const float* b2a_b2 = (const float*)d_in[27];
    const float* b2b_w1 = (const float*)d_in[28]; const float* b2b_s1 = (const float*)d_in[29]; const float* b2b_b1 = (const float*)d_in[30];
    const float* b2b_w2 = (const float*)d_in[31]; const float* b2b_s2 = (const float*)d_in[32]; const float* b2b_b2 = (const float*)d_in[33];
    const float* d3_w = (const float*)d_in[34]; const float* d3_s = (const float*)d_in[35]; const float* d3_b = (const float*)d_in[36];
    const float* b3a_w1 = (const float*)d_in[37]; const float* b3a_s1 = (const float*)d_in[38]; const float* b3a_b1 = (const float*)d_in[39];
    const float* b3a_w2 = (const float*)d_in[40]; const float* b3a_s2 = (const float*)d_in[41]; const float* b3a_b2 = (const float*)d_in[42];
    const float* b3b_w1 = (const float*)d_in[43]; const float* b3b_s1 = (const float*)d_in[44]; const float* b3b_b1 = (const float*)d_in[45];
    const float* b3b_w2 = (const float*)d_in[46]; const float* b3b_s2 = (const float*)d_in[47]; const float* b3b_b2 = (const float*)d_in[48];
    const float* d4_w = (const float*)d_in[49]; const float* d4_s = (const float*)d_in[50]; const float* d4_b = (const float*)d_in[51];
    const float* b4a_w1 = (const float*)d_in[52]; const float* b4a_s1 = (const float*)d_in[53]; const float* b4a_b1 = (const float*)d_in[54];
    const float* b4a_w2 = (const float*)d_in[55]; const float* b4a_s2 = (const float*)d_in[56]; const float* b4a_b2 = (const float*)d_in[57];
    const float* b4b_w1 = (const float*)d_in[58]; const float* b4b_s1 = (const float*)d_in[59]; const float* b4b_b1 = (const float*)d_in[60];
    const float* b4b_w2 = (const float*)d_in[61]; const float* b4b_s2 = (const float*)d_in[62]; const float* b4b_b2 = (const float*)d_in[63];
    const float* ex_w = (const float*)d_in[64]; const float* ex_s = (const float*)d_in[65]; const float* ex_b = (const float*)d_in[66];

    // Fixed shapes from setup_inputs(): B=2, grid (D=41,H=64,W=64).
    const int Bn = 2;
    const int D1 = 41, H1 = 64, W1 = 64;  const int NV1 = Bn * D1 * H1 * W1;   // 335872
    const int D2 = 21, H2 = 32, W2 = 32;  const int NV2 = Bn * D2 * H2 * W2;   // 43008
    const int D3 = 11, H3 = 16, W3 = 16;  const int NV3 = Bn * D3 * H3 * W3;   // 5632
    const int D4 = 5,  H4 = 8,  W4 = 8;   const int NV4 = Bn * D4 * H4 * W4;   // 640
    const int D5 = 2,  H5 = 8,  W5 = 8;   const int NV5 = Bn * D5 * H5 * W5;   // 256
    const int C0p = 96;                    // 65 channels padded to 3*32

    // --- workspace carve-out ---
    char* ws = (char*)d_ws;
    size_t off = 0;
    auto alloc = [&](size_t bytes) -> char* {
        char* p = ws + off;
        off = (off + bytes + 255) & ~(size_t)255;
        return p;
    };
    unsigned short* zp  = (unsigned short*)alloc(256);          // zero page
    unsigned short* g0  = (unsigned short*)alloc((size_t)NV1 * C0p * 2);
    float*          m0  = (float*)alloc((size_t)NV1 * 4);
    unsigned short* a1a = (unsigned short*)alloc((size_t)NV1 * 32 * 2);
    unsigned short* a1b = (unsigned short*)alloc((size_t)NV1 * 32 * 2);
    float*          m2  = (float*)alloc((size_t)NV2 * 4);
    unsigned short* a2a = (unsigned short*)alloc((size_t)NV2 * 32 * 2);
    unsigned short* a2b = (unsigned short*)alloc((size_t)NV2 * 32 * 2);
    float*          m3  = (float*)alloc((size_t)NV3 * 4);
    unsigned short* a3a = (unsigned short*)alloc((size_t)NV3 * 64 * 2);
    unsigned short* a3b = (unsigned short*)alloc((size_t)NV3 * 64 * 2);
    float*          m4  = (float*)alloc((size_t)NV4 * 4);
    unsigned short* a4a = (unsigned short*)alloc((size_t)NV4 * 128 * 2);
    unsigned short* a4b = (unsigned short*)alloc((size_t)NV4 * 128 * 2);
    float*          m5  = (float*)alloc((size_t)NV5 * 4);
    unsigned short* exo = (unsigned short*)alloc((size_t)NV5 * 128 * 2);

    auto psize = [](int taps, int Cinp, int Cout) -> size_t {
        return (size_t)taps * (Cinp / 32) * (Cout / 16) * 512 * 2;
    };
    unsigned short* p_in  = (unsigned short*)alloc(psize(27, 96, 16));
    unsigned short* p1a1  = (unsigned short*)alloc(psize(27, 32, 16));
    unsigned short* p1a2  = (unsigned short*)alloc(psize(27, 32, 16));
    unsigned short* p1b1  = (unsigned short*)alloc(psize(27, 32, 16));
    unsigned short* p1b2  = (unsigned short*)alloc(psize(27, 32, 16));
    unsigned short* pd2   = (unsigned short*)alloc(psize(27, 32, 32));
    unsigned short* p2a1  = (unsigned short*)alloc(psize(27, 32, 32));
    unsigned short* p2a2  = (unsigned short*)alloc(psize(27, 32, 32));
    unsigned short* p2b1  = (unsigned short*)alloc(psize(27, 32, 32));
    unsigned short* p2b2  = (unsigned short*)alloc(psize(27, 32, 32));
    unsigned short* pd3   = (unsigned short*)alloc(psize(27, 32, 64));
    unsigned short* p3a1  = (unsigned short*)alloc(psize(27, 64, 64));
    unsigned short* p3a2  = (unsigned short*)alloc(psize(27, 64, 64));
    unsigned short* p3b1  = (unsigned short*)alloc(psize(27, 64, 64));
    unsigned short* p3b2  = (unsigned short*)alloc(psize(27, 64, 64));
    unsigned short* pd4   = (unsigned short*)alloc(psize(27, 64, 128));
    unsigned short* p4a1  = (unsigned short*)alloc(psize(27, 128, 128));
    unsigned short* p4a2  = (unsigned short*)alloc(psize(27, 128, 128));
    unsigned short* p4b1  = (unsigned short*)alloc(psize(27, 128, 128));
    unsigned short* p4b2  = (unsigned short*)alloc(psize(27, 128, 128));
    unsigned short* pex   = (unsigned short*)alloc(psize(3, 128, 128));

    // --- zero: zero page, scatter grid, mask, stage-1 buffers (pad channels) ---
    auto zero = [&](void* p, size_t bytes) {
        long n = (long)(bytes / 4);
        int blocks = (int)((n + 255) / 256);
        if (blocks > 4096) blocks = 4096;
        zero_u32<<<blocks, 256, 0, stream>>>((unsigned int*)p, n);
    };
    zero(zp,  256);
    zero(g0,  (size_t)NV1 * C0p * 2);
    zero(m0,  (size_t)NV1 * 4);
    zero(a1a, (size_t)NV1 * 32 * 2);
    zero(a1b, (size_t)NV1 * 32 * 2);

    // --- weight repack ---
    auto pack = [&](const float* W, unsigned short* dst, int taps, int Cin, int Cinp, int Cout) {
        long total = (long)taps * (Cinp / 32) * (Cout / 16) * 512;
        pack_weights<<<(int)((total + 255) / 256), 256, 0, stream>>>(W, dst, taps, Cin, Cinp, Cout);
    };
    pack(W_in,   p_in, 27, 65, 96, 16);
    pack(b1a_w1, p1a1, 27, 16, 32, 16);  pack(b1a_w2, p1a2, 27, 16, 32, 16);
    pack(b1b_w1, p1b1, 27, 16, 32, 16);  pack(b1b_w2, p1b2, 27, 16, 32, 16);
    pack(d2_w,   pd2,  27, 16, 32, 32);
    pack(b2a_w1, p2a1, 27, 32, 32, 32);  pack(b2a_w2, p2a2, 27, 32, 32, 32);
    pack(b2b_w1, p2b1, 27, 32, 32, 32);  pack(b2b_w2, p2b2, 27, 32, 32, 32);
    pack(d3_w,   pd3,  27, 32, 32, 64);
    pack(b3a_w1, p3a1, 27, 64, 64, 64);  pack(b3a_w2, p3a2, 27, 64, 64, 64);
    pack(b3b_w1, p3b1, 27, 64, 64, 64);  pack(b3b_w2, p3b2, 27, 64, 64, 64);
    pack(d4_w,   pd4,  27, 64, 64, 128);
    pack(b4a_w1, p4a1, 27, 128, 128, 128); pack(b4a_w2, p4a2, 27, 128, 128, 128);
    pack(b4b_w1, p4b1, 27, 128, 128, 128); pack(b4b_w2, p4b2, 27, 128, 128, 128);
    pack(ex_w,   pex,   3, 128, 128, 128);

    // --- embed + scatter ---
    embed_scatter<<<(N + 255) / 256, 256, 0, stream>>>(vf, vc, g0, m0, N, D1, H1, W1, C0p);

    // Stage 1: input conv + two subm blocks @16ch (stored padded to 32).
    launch_conv<4,1>(stream, g0,  p_in, S_in,   B_in,   m0, nullptr, zp, a1a, Bn, D1,H1,W1,96, D1,H1,W1,32,16, 3,3,3, 1,1,1, 1,1,1);
    launch_conv<4,1>(stream, a1a, p1a1, b1a_s1, b1a_b1, m0, nullptr, zp, a1b, Bn, D1,H1,W1,32, D1,H1,W1,32,16, 3,3,3, 1,1,1, 1,1,1);
    launch_conv<4,1>(stream, a1b, p1a2, b1a_s2, b1a_b2, m0, a1a,     zp, a1a, Bn, D1,H1,W1,32, D1,H1,W1,32,16, 3,3,3, 1,1,1, 1,1,1);
    launch_conv<4,1>(stream, a1a, p1b1, b1b_s1, b1b_b1, m0, nullptr, zp, a1b, Bn, D1,H1,W1,32, D1,H1,W1,32,16, 3,3,3, 1,1,1, 1,1,1);
    launch_conv<4,1>(stream, a1b, p1b2, b1b_s2, b1b_b2, m0, a1a,     zp, a1a, Bn, D1,H1,W1,32, D1,H1,W1,32,16, 3,3,3, 1,1,1, 1,1,1);

    // Stage 2: downsample to 32ch.
    mask_down<<<(NV2 + 255) / 256, 256, 0, stream>>>(m0, m2, Bn, D1,H1,W1, D2,H2,W2, 3,3,3, 2,2,2, 1,1,1);
    launch_conv<4,2>(stream, a1a, pd2,  d2_s,   d2_b,   m2, nullptr, zp, a2a, Bn, D1,H1,W1,32, D2,H2,W2,32,32, 3,3,3, 2,2,2, 1,1,1);
    launch_conv<4,2>(stream, a2a, p2a1, b2a_s1, b2a_b1, m2, nullptr, zp, a2b, Bn, D2,H2,W2,32, D2,H2,W2,32,32, 3,3,3, 1,1,1, 1,1,1);
    launch_conv<4,2>(stream, a2b, p2a2, b2a_s2, b2a_b2, m2, a2a,     zp, a2a, Bn, D2,H2,W2,32, D2,H2,W2,32,32, 3,3,3, 1,1,1, 1,1,1);
    launch_conv<4,2>(stream, a2a, p2b1, b2b_s1, b2b_b1, m2, nullptr, zp, a2b, Bn, D2,H2,W2,32, D2,H2,W2,32,32, 3,3,3, 1,1,1, 1,1,1);
    launch_conv<4,2>(stream, a2b, p2b2, b2b_s2, b2b_b2, m2, a2a,     zp, a2a, Bn, D2,H2,W2,32, D2,H2,W2,32,32, 3,3,3, 1,1,1, 1,1,1);

    // Stage 3: downsample to 64ch.
    mask_down<<<(NV3 + 255) / 256, 256, 0, stream>>>(m2, m3, Bn, D2,H2,W2, D3,H3,W3, 3,3,3, 2,2,2, 1,1,1);
    launch_conv<4,2>(stream, a2a, pd3,  d3_s,   d3_b,   m3, nullptr, zp, a3a, Bn, D2,H2,W2,32, D3,H3,W3,64,64, 3,3,3, 2,2,2, 1,1,1);
    launch_conv<4,2>(stream, a3a, p3a1, b3a_s1, b3a_b1, m3, nullptr, zp, a3b, Bn, D3,H3,W3,64, D3,H3,W3,64,64, 3,3,3, 1,1,1, 1,1,1);
    launch_conv<4,2>(stream, a3b, p3a2, b3a_s2, b3a_b2, m3, a3a,     zp, a3a, Bn, D3,H3,W3,64, D3,H3,W3,64,64, 3,3,3, 1,1,1, 1,1,1);
    launch_conv<4,2>(stream, a3a, p3b1, b3b_s1, b3b_b1, m3, nullptr, zp, a3b, Bn, D3,H3,W3,64, D3,H3,W3,64,64, 3,3,3, 1,1,1, 1,1,1);
    launch_conv<4,2>(stream, a3b, p3b2, b3b_s2, b3b_b2, m3, a3a,     zp, a3a, Bn, D3,H3,W3,64, D3,H3,W3,64,64, 3,3,3, 1,1,1, 1,1,1);

    // Stage 4: downsample to 128ch, pad (0,1,1).
    mask_down<<<(NV4 + 255) / 256, 256, 0, stream>>>(m3, m4, Bn, D3,H3,W3, D4,H4,W4, 3,3,3, 2,2,2, 0,1,1);
    launch_conv<4,2>(stream, a3a, pd4,  d4_s,   d4_b,   m4, nullptr, zp, a4a, Bn, D3,H3,W3,64, D4,H4,W4,128,128, 3,3,3, 2,2,2, 0,1,1);
    launch_conv<4,2>(stream, a4a, p4a1, b4a_s1, b4a_b1, m4, nullptr, zp, a4b, Bn, D4,H4,W4,128, D4,H4,W4,128,128, 3,3,3, 1,1,1, 1,1,1);
    launch_conv<4,2>(stream, a4b, p4a2, b4a_s2, b4a_b2, m4, a4a,     zp, a4a, Bn, D4,H4,W4,128, D4,H4,W4,128,128, 3,3,3, 1,1,1, 1,1,1);
    launch_conv<4,2>(stream, a4a, p4b1, b4b_s1, b4b_b1, m4, nullptr, zp, a4b, Bn, D4,H4,W4,128, D4,H4,W4,128,128, 3,3,3, 1,1,1, 1,1,1);
    launch_conv<4,2>(stream, a4b, p4b2, b4b_s2, b4b_b2, m4, a4a,     zp, a4a, Bn, D4,H4,W4,128, D4,H4,W4,128,128, 3,3,3, 1,1,1, 1,1,1);

    // Extra layer: (3,1,1) conv, stride (2,1,1), no pad.
    mask_down<<<(NV5 + 255) / 256, 256, 0, stream>>>(m4, m5, Bn, D4,H4,W4, D5,H5,W5, 3,1,1, 2,1,1, 0,0,0);
    launch_conv<4,2>(stream, a4a, pex, ex_s, ex_b, m5, nullptr, zp, exo, Bn, D4,H4,W4,128, D5,H5,W5,128,128, 3,1,1, 2,1,1, 0,0,0);

    // Final layout transform to f32 output.
    final_transpose<<<(2 * 256 * 8 * 8 + 255) / 256, 256, 0, stream>>>(exo, (float*)d_out);
}
